// ProjectAndSampleGrid_81260781240375
// MI455X (gfx1250) — matrix operand order repaired and checked by reading, not compiled, not run on hardware
//
#include <hip/hip_runtime.h>

// ProjectAndSampleGrid for MI455X (gfx1250, wave32).
// Wave tile: 32 consecutive output j (fixed b,z,i) x 32 channels.
// Bilinear sum done with v_wmma_f32_16x16x32_f16:
//   D(16 pts x 16 ch) = A(16x32, block-diag weights) @ B(32x16, gathered corners)
// f16 weights/values, f32 accumulate. Stores: one full 128B line per (wave, channel).
// All gathers/stores use wave-uniform base + 32-bit lane index -> saddr addressing.

typedef __attribute__((ext_vector_type(16))) _Float16 v16h;
typedef __attribute__((ext_vector_type(8)))  float    v8f;
typedef __attribute__((ext_vector_type(8)))  unsigned v8u;

#define C_CH 32
#define IH   135
#define IW   240
#define HWSZ (IH * IW)
#define OH   560   // i axis
#define OW   960   // j axis

static __device__ __forceinline__ unsigned pkh(float a, float b) {
#if __has_builtin(__builtin_amdgcn_cvt_pkrtz)
    typedef __fp16 h2_t __attribute__((ext_vector_type(2)));
    h2_t p = __builtin_amdgcn_cvt_pkrtz(a, b);
    return __builtin_bit_cast(unsigned, p);
#else
    union { _Float16 h[2]; unsigned u; } v;
    v.h[0] = (_Float16)a; v.h[1] = (_Float16)b;
    return v.u;
#endif
}

__global__ __launch_bounds__(256)
void ProjectAndSampleGrid_81260781240375_kernel(const float* __restrict__ feat,
                                                const float* __restrict__ dist_poly,
                                                const float* __restrict__ cam,
                                                float* __restrict__ out)
{
    const int lane = threadIdx.x;          // 0..31
    const int wv   = threadIdx.y;          // 0..7  (wave in block)
    const int bz   = blockIdx.z;           // 0..7  (b*4+z)
    const int b    = bz >> 2;
    const int z    = bz & 3;
    const int i    = blockIdx.y * 8 + wv;  // 0..559
    const int jb   = blockIdx.x * 32;      // 0..928

    // per-wave point table: 4 gather element-offsets + 4 masked bilinear weights
    __shared__ int   soff[8][32][4];
    __shared__ float swt [8][32][4];

    // ---------------- phase 1: per-lane projection of point j = jb+lane ----
    {
        const int j = jb + lane;
        // OUT[b,z,c,i,j] = S[bz,c, gy=959-j, gx=559-i]; gnd[r,cc] = (cc/8-35, r/8-60)
        const float X = (float)(559 - i) * 0.125f - 35.0f;
        const float Y = (float)(959 - j) * 0.125f - 60.0f;
        const float Z = (2.0f / 3.0f) * (float)z;     // linspace(0,2,4)

        const float* cm = cam + b * 12;
        const float px = cm[0]*X + cm[1]*Y + cm[2]*Z  + cm[3];
        const float py = cm[4]*X + cm[5]*Y + cm[6]*Z  + cm[7];
        const float pz = cm[8]*X + cm[9]*Y + cm[10]*Z + cm[11];

        const float sgn = (pz >= 0.0f) ? 1.0f : -1.0f;
        const float den = (fabsf(pz) >= 1.0e-4f) ? pz : sgn * 1.0e-4f;
        float ux = px / den;
        float uy = py / den;
        ux = (ux != ux) ? 0.0f : (isinf(ux) ? copysignf(65504.0f, ux) : ux);
        uy = (uy != uy) ? 0.0f : (isinf(uy) ? copysignf(65504.0f, uy) : uy);

        const float rr = sqrtf(ux*ux + uy*uy);
        const float rs = fmaxf(rr, 1.0e-4f);
        const float th = atanf(rs);

        const float* pp = dist_poly + b * 9;
        float s = th * pp[0] + pp[1];
        #pragma unroll
        for (int k = 2; k < 9; ++k) s = th * s + pp[k];
        const float scl = s / rs;

        const float gx  = ux * scl;
        const float gy  = uy * scl;
        const float sgx = (7680.0f * gx) / 3839.0f;   // 2*IMG_W/(IMG_W-1)
        const float sgy = (7680.0f * gy) / 2159.0f;   // 2*IMG_W/(IMG_H-1)
        const float fx  = (sgx + 1.0f) * 120.0f - 0.5f;   // Wi=240
        const float fy  = (sgy + 1.0f) * 67.5f  - 0.5f;   // Hi=135

        const float x0f = floorf(fx);
        const float y0f = floorf(fy);
        const float wx  = fx - x0f;
        const float wy  = fy - y0f;

        const float vx0 = (x0f >=  0.0f && x0f <= 239.0f) ? 1.0f : 0.0f;
        const float vx1 = (x0f >= -1.0f && x0f <= 238.0f) ? 1.0f : 0.0f;
        const float vy0 = (y0f >=  0.0f && y0f <= 134.0f) ? 1.0f : 0.0f;
        const float vy1 = (y0f >= -1.0f && y0f <= 133.0f) ? 1.0f : 0.0f;

        const int x0c = (int)fminf(fmaxf(x0f,        0.0f), 239.0f);
        const int x1c = (int)fminf(fmaxf(x0f + 1.0f, 0.0f), 239.0f);
        const int y0c = (int)fminf(fmaxf(y0f,        0.0f), 134.0f);
        const int y1c = (int)fminf(fmaxf(y0f + 1.0f, 0.0f), 134.0f);

        soff[wv][lane][0] = y0c * IW + x0c;
        soff[wv][lane][1] = y0c * IW + x1c;
        soff[wv][lane][2] = y1c * IW + x0c;
        soff[wv][lane][3] = y1c * IW + x1c;
        swt [wv][lane][0] = (1.0f - wx) * (1.0f - wy) * vx0 * vy0;
        swt [wv][lane][1] = wx * (1.0f - wy) * vx1 * vy0;
        swt [wv][lane][2] = (1.0f - wx) * wy * vx0 * vy1;
        swt [wv][lane][3] = wx * wy * vx1 * vy1;
    }
    __syncthreads();

    const int hi = lane >> 4;
    const int cl = lane & 15;
    // wave-uniform bases; per-lane access via 32-bit element index (saddr form)
    const float* featU = feat + (size_t)b * (C_CH * HWSZ);     // z-broadcast feature map
    float*       outU  = out  + (size_t)bz * (C_CH * OH * OW);

    // K assignment (shared by A and B):
    //   q' = point-within-8; holder side hi_A = q'>>2; A regs vb=2*(q'&3), vb+1.
    //   B lane hi: element group r (4 f16) holds point q' = r|(hi<<2).
    #pragma unroll
    for (int grp = 0; grp < 2; ++grp) {
        v16h A0, A1;
        {
            const int  m    = cl;            // D row held by this lane pair
            const int  q    = m & 7;
            const int  vb   = 2 * (q & 3);
            const bool side = (hi == (q >> 2));
            const float w0 = swt[wv][grp*16 + m][0];
            const float w1 = swt[wv][grp*16 + m][1];
            const float w2 = swt[wv][grp*16 + m][2];
            const float w3 = swt[wv][grp*16 + m][3];
            const unsigned P0 = pkh(w0, w1);
            const unsigned P1 = pkh(w2, w3);
            const bool act0 = ((m >> 3) == 0) && side;   // WMMA t=0 covers rows 0-7
            const bool act1 = ((m >> 3) == 1) && side;   // WMMA t=1 covers rows 8-15
            const unsigned Q0a = act0 ? P0 : 0u, Q1a = act0 ? P1 : 0u;
            const unsigned Q0b = act1 ? P0 : 0u, Q1b = act1 ? P1 : 0u;
            v8u a0u, a1u;
            #pragma unroll
            for (int r = 0; r < 8; ++r) {
                a0u[r] = (r == vb) ? Q0a : ((r == vb + 1) ? Q1a : 0u);
                a1u[r] = (r == vb) ? Q0b : ((r == vb + 1) ? Q1b : 0u);
            }
            A0 = __builtin_bit_cast(v16h, a0u);
            A1 = __builtin_bit_cast(v16h, a1u);
        }

        #pragma unroll
        for (int half = 0; half < 2; ++half) {
            const int c  = half * 16 + cl;       // this lane's channel (B column)
            const int co = c * HWSZ;             // 32-bit channel base (element units)
            v8f acc = {};
            #pragma unroll
            for (int t = 0; t < 2; ++t) {
                v8u bu;
                #pragma unroll
                for (int r = 0; r < 4; ++r) {
                    const int q = r | (hi << 2);
                    const int P = grp * 16 + t * 8 + q;   // point index in wave
                    const float f0 = featU[co + soff[wv][P][0]];
                    const float f1 = featU[co + soff[wv][P][1]];
                    const float f2 = featU[co + soff[wv][P][2]];
                    const float f3 = featU[co + soff[wv][P][3]];
                    bu[2*r]     = pkh(f0, f1);
                    bu[2*r + 1] = pkh(f2, f3);
                }
                const v16h bm = __builtin_bit_cast(v16h, bu);
                acc = __builtin_amdgcn_wmma_f32_16x16x32_f16(
                          false, (t == 0) ? A0 : A1, false, bm,
                          (short)0, acc, false, false);
            }
            // D VGPR g holds row M = g + 8*hi -> j = jb + grp*16 + 8*hi + g (contiguous)
            const int j0  = jb + grp * 16 + hi * 8;
            const int oix = (c * OH + i) * OW + j0;      // < 17.2M, fits 32-bit
            float4 s0 = make_float4(acc[0], acc[1], acc[2], acc[3]);
            float4 s1 = make_float4(acc[4], acc[5], acc[6], acc[7]);
            *reinterpret_cast<float4*>(outU + oix)     = s0;
            *reinterpret_cast<float4*>(outU + oix + 4) = s1;
        }
    }
}

extern "C" void kernel_launch(void* const* d_in, const int* in_sizes, int n_in,
                              void* d_out, int out_size, void* d_ws, size_t ws_size,
                              hipStream_t stream) {
    (void)in_sizes; (void)n_in; (void)d_ws; (void)ws_size; (void)out_size;
    const float* feat = (const float*)d_in[0];   // (2,32,135,240) f32
    const float* poly = (const float*)d_in[1];   // (2,9) f32
    const float* cm   = (const float*)d_in[2];   // (2,3,4) f32
    float* out = (float*)d_out;                  // (2,4,32,560,960) f32

    dim3 block(32, 8, 1);                        // 8 wave32 per block
    dim3 grid(OW / 32, OH / 8, 8);               // (30, 70, 8)
    hipLaunchKernelGGL(ProjectAndSampleGrid_81260781240375_kernel,
                       grid, block, 0, stream, feat, poly, cm, out);
}
